// MultiheadAttention_77953656422520
// MI455X (gfx1250) — matrix operand order repaired and checked by reading, not compiled, not run on hardware
//
#include <hip/hip_runtime.h>

#define E_DIM 1024
#define HEADS 16
#define HD 64
#define BB 2
#define QL 2048
#define KL 2048

typedef __attribute__((ext_vector_type(16))) __bf16 v16bf;
typedef __attribute__((ext_vector_type(8)))  float  v8f;
typedef __attribute__((ext_vector_type(4)))  __bf16 v4bf;
typedef unsigned int u32;
typedef __attribute__((ext_vector_type(4))) u32 u32x4;
typedef __attribute__((ext_vector_type(8))) u32 u32x8;

union FragU { v16bf v; uint4 u[2]; };

__device__ __forceinline__ v8f v8f_zero() {
  v8f z;
#pragma unroll
  for (int r = 0; r < 8; ++r) z[r] = 0.0f;
  return z;
}

__device__ __forceinline__ v8f wmma_bf16(v16bf a, v16bf b, v8f c) {
  // D = A(16x32 bf16) x B(32x16 bf16) + C(16x16 f32)
  return __builtin_amdgcn_wmma_f32_16x16x32_bf16(false, a, false, b, (short)0, c,
                                                 false, false);
}

// A fragment 16x32 (MxK), row-major A. Contiguous k order (B matches): lane m = lane&15,
// half h = lane>>4, element e -> k = 16*h + e.
// => one contiguous 32B chunk per lane at k0+16h.
__device__ __forceinline__ v16bf load_A_frag(const __bf16* A, int lda, int row0,
                                             int k0, int lane) {
  const int m = lane & 15, h = lane >> 4;
  const __bf16* p = A + (size_t)(row0 + m) * lda + k0 + 16 * h;
  FragU f;
  f.u[0] = *(const uint4*)p;
  f.u[1] = *(const uint4*)(p + 8);
  return f.v;
}

// B fragment 32x16 (KxN) for NT GEMM: memory holds B as rows of length K
// (B[n][k]). Same contiguous k order as A: lane n = lane&15, half h = lane>>4,
// element e -> k = 16*h + e  => one contiguous 32B chunk per lane.
__device__ __forceinline__ v16bf load_B_frag(const __bf16* Bm, int ldb, int col0,
                                             int k0, int lane) {
  const int n = lane & 15, h = lane >> 4;
  const __bf16* p = Bm + (size_t)(col0 + n) * ldb + k0 + 16 * h;
  FragU f;
  f.u[0] = *(const uint4*)p;
  f.u[1] = *(const uint4*)(p + 8);
  return f.v;
}

// gfx1250 async global->LDS copy (ASYNCcnt-tracked), 16B per lane.
__device__ __forceinline__ void async_copy_b128(const __bf16* lds_dst,
                                                const __bf16* gsrc) {
  unsigned int loff = (unsigned int)(uintptr_t)lds_dst;
  asm volatile("global_load_async_to_lds_b128 %0, %1, off"
               :
               : "v"(loff), "v"(gsrc)
               : "memory");
}

__device__ __forceinline__ void wait_asynccnt0() {
  asm volatile("s_wait_asynccnt 0x0" ::: "memory");
}

// Tensor Data Mover: 2-D tile load Global->LDS (TENSORcnt-tracked).
// Builds D# group0 (4 SGPRs) + group1 (8 SGPRs) per CDNA5 ISA ch.8;
// VADDR2/3 omitted (2-D tensor). data_size = 2 bytes (bf16).
// tile0 = elements per row, tile1 = rows, stride = row stride in elements.
__device__ __forceinline__ void tdm_load_2d(const void* gptr, u32 lds_addr,
                                            u32 tile0, u32 tile1, u32 stride) {
  const unsigned long long ga = (unsigned long long)(uintptr_t)gptr;
  const u32 galo = __builtin_amdgcn_readfirstlane((u32)ga);
  const u32 gahi = __builtin_amdgcn_readfirstlane((u32)(ga >> 32));
  const u32 laddr = __builtin_amdgcn_readfirstlane(lds_addr);
  const u32 TD = 1u << 20;  // tensor_dim0/1: OOB bound only, tiles stay inside
  u32x4 g0;
  g0[0] = 1u;                                  // count=1, user descriptor
  g0[1] = laddr;                               // lds_addr (bytes)
  g0[2] = galo;                                // global_addr[31:0]
  g0[3] = (gahi & 0x01FFFFFFu) | (2u << 30);   // global_addr[56:32] | type=2
  u32x8 g1;
  g1[0] = 0x00010000u;                         // wg_mask=0, data_size=1 (2B)
  g1[1] = (TD & 0xFFFFu) << 16;                              // tensor_dim0 lo
  g1[2] = ((TD >> 16) & 0xFFFFu) | ((TD & 0xFFFFu) << 16);   // td0 hi | td1 lo
  g1[3] = ((TD >> 16) & 0xFFFFu) | (tile0 << 16);            // td1 hi | tile0
  g1[4] = tile1;                               // tile_dim1 | tile_dim2=0
  g1[5] = stride;                              // tensor_dim0_stride[31:0]
  g1[6] = 0u;                                  // stride[47:32] | dim1_stride lo
  g1[7] = 0u;
  asm volatile("tensor_load_to_lds %0, %1" : : "s"(g0), "s"(g1) : "memory");
}

// ---------------------------------------------------------------- conversion
__global__ __launch_bounds__(256) void cvt_f32_to_bf16(const float* __restrict__ in,
                                                       __bf16* __restrict__ out,
                                                       int n4) {
  int i = blockIdx.x * blockDim.x + threadIdx.x;
  if (i < n4) {
    const float4 f = ((const float4*)in)[i];
    v4bf o;
    o.x = (__bf16)f.x; o.y = (__bf16)f.y; o.z = (__bf16)f.z; o.w = (__bf16)f.w;
    ((v4bf*)out)[i] = o;
  }
}

// ----------------------------------------------------------------- NT GEMM
// C[M,N] = A[M,K] @ B[N,K]^T, bf16 in, f32 accumulate.
// EPI 0: f32 row-major out. EPI 1: bf16 row-major out.
// EPI 2: bf16 per-head transposed out (V^T): [b,h][d][kpos], kpos stride 2048.
// Block = 4 waves (128 thr); wave tile 32x64; block tile 128x64.
template <int EPI>
__global__ __launch_bounds__(128)
void gemm_nt_bf16(const __bf16* __restrict__ A, const __bf16* __restrict__ Bm,
                  void* __restrict__ Cout, int M, int N, int K) {
  const int lane = threadIdx.x & 31;
  const int w = threadIdx.x >> 5;
  const int n0 = blockIdx.x * 64;
  const int m0 = blockIdx.y * 128 + w * 32;
  const int nl = lane & 15, hh = lane >> 4;

  v8f acc[2][4];
#pragma unroll
  for (int i = 0; i < 2; ++i)
#pragma unroll
    for (int j = 0; j < 4; ++j) acc[i][j] = v8f_zero();

  for (int k0 = 0; k0 < K; k0 += 32) {
    v16bf a0 = load_A_frag(A, K, m0, k0, lane);
    v16bf a1 = load_A_frag(A, K, m0 + 16, k0, lane);
    if (k0 + 64 < K) {  // hint next K-step into cache (global_prefetch_b8)
      __builtin_prefetch(A + (size_t)(m0 + nl) * K + k0 + 64, 0, 0);
      __builtin_prefetch(Bm + (size_t)(n0 + nl) * K + k0 + 64, 0, 0);
    }
#pragma unroll
    for (int j = 0; j < 4; ++j) {
      v16bf b = load_B_frag(Bm, K, n0 + j * 16, k0, lane);
      acc[0][j] = wmma_bf16(a0, b, acc[0][j]);
      acc[1][j] = wmma_bf16(a1, b, acc[1][j]);
    }
  }

  // C layout: element r of acc vector is row m0+i*16 + r + 8*hh, col n0+j*16+nl
#pragma unroll
  for (int i = 0; i < 2; ++i)
#pragma unroll
    for (int j = 0; j < 4; ++j)
#pragma unroll
      for (int r = 0; r < 8; ++r) {
        const int gm = m0 + i * 16 + r + 8 * hh;
        const int gn = n0 + j * 16 + nl;
        const float v = acc[i][j][r];
        if (EPI == 0) {
          ((float*)Cout)[(size_t)gm * N + gn] = v;
        } else if (EPI == 1) {
          ((__bf16*)Cout)[(size_t)gm * N + gn] = (__bf16)v;
        } else {
          const int bb = gm >> 11, kp = gm & 2047;   // 2048 rows per batch
          const int hd = gn >> 6,  dd = gn & 63;     // head, dim
          ((__bf16*)Cout)[(((size_t)(bb * HEADS + hd) * HD + dd) << 11) + kp] =
              (__bf16)v;
        }
      }
}

// ----------------------------------------------------------- fused attention
// One block (8 waves) per (b, h, 128-q-row slab). All waves share 64-key
// K/V tiles, double-buffered in LDS: K tile staged by one TDM
// tensor_load_to_lds (TENSORcnt), V tile by per-lane async copies (ASYNCcnt).
// 18 WMMAs per 64-key step: 8 logits, 2 row-sum (P @ ones), 8 P@V.
__global__ __launch_bounds__(256)
void attn_fused(const __bf16* __restrict__ Qb, const __bf16* __restrict__ Kb,
                const __bf16* __restrict__ Vt, const int* __restrict__ mask,
                __bf16* __restrict__ O) {
  __shared__ __bf16 kTile[2][64 * HD];  // [key][d]    8KB per buffer
  __shared__ __bf16 vTile[2][HD * 64];  // [d][kpos]   8KB per buffer
  __shared__ __bf16 ldsP[8][16 * 64];   // per-wave P staging (C->A relayout)

  const int tid = threadIdx.x;
  const int lane = tid & 31;
  const int w = tid >> 5;
  const int pair = blockIdx.x >> 4;  // 32 (b,h) pairs
  const int qblk = blockIdx.x & 15;  // 16 slabs of 128 q rows
  const int h = pair & 15;
  const int b = pair >> 4;
  const int qt = qblk * 8 + w;       // this wave's 16-row q tile
  const int nl = lane & 15, hh = lane >> 4;

  const __bf16* Qbase = Qb + ((size_t)(b * QL + qt * 16)) * E_DIM + h * HD;
  const __bf16* Kbase = Kb + (size_t)b * KL * E_DIM + h * HD;
  const __bf16* Vbase = Vt + ((size_t)(b * HEADS + h)) * HD * KL;
  const int* mrow = mask + b * KL;

  // Q tile fragments kept resident: 16 rows x 64 dims = two 16x32 A frags
  const v16bf aq0 = load_A_frag(Qbase, E_DIM, 0, 0, lane);
  const v16bf aq1 = load_A_frag(Qbase, E_DIM, 0, 32, lane);

  v16bf ones;
#pragma unroll
  for (int i = 0; i < 16; ++i) ones[i] = (__bf16)1.0f;

  v8f o0 = v8f_zero(), o1 = v8f_zero(), o2 = v8f_zero(), o3 = v8f_zero();
  float rmax[8], rsum[8];
#pragma unroll
  for (int r = 0; r < 8; ++r) { rmax[r] = -3.0e38f; rsum[r] = 0.0f; }

  const float scale = 0.03125f;  // 1/sqrt(1024)

  auto stageK = [&](int kb, int bufi) {  // one TDM op: 64 rows x 128B
    tdm_load_2d(Kbase + (size_t)kb * E_DIM, (u32)(uintptr_t)&kTile[bufi][0],
                HD, 64, E_DIM);
  };
  auto stageV = [&](int kb, int bufi) {  // 512 x 16B chunks, 2 per thread
#pragma unroll
    for (int i = 0; i < 2; ++i) {
      const int cid = tid + i * 256;
      const int r = cid >> 3, c = (cid & 7) * 8;
      async_copy_b128(&vTile[bufi][r * 64 + c], Vbase + (size_t)r * KL + kb + c);
    }
  };

  if (w == 0) stageK(0, 0);
  stageV(0, 0);
  int buf = 0;

  for (int kb = 0; kb < KL; kb += 64) {
    if (w == 0) __builtin_amdgcn_s_wait_tensorcnt(0);  // K tile landed (TDM)
    wait_asynccnt0();                                  // V chunks landed
    __syncthreads();  // everyone's copies visible; prior reads of buf done
    if (kb + 64 < KL) {
      if (w == 0) stageK(kb + 64, buf ^ 1);
      stageV(kb + 64, buf ^ 1);
    }

    // ---- logits for 64 keys: four 16x16 tiles, each = 2 WMMAs over d=64
    float l[4][8];
#pragma unroll
    for (int t = 0; t < 4; ++t) {
      v8f c = v8f_zero();
      c = wmma_bf16(aq0, load_B_frag(kTile[buf], HD, 16 * t, 0, lane), c);
      c = wmma_bf16(aq1, load_B_frag(kTile[buf], HD, 16 * t, 32, lane), c);
      const float mv = 10000.0f * (float)mrow[kb + 16 * t + nl];
#pragma unroll
      for (int r = 0; r < 8; ++r) l[t][r] = c[r] * scale - mv;
    }

    // ---- online softmax; row m = r + 8*hh lives across 16 lanes of a half
    float tmax[8];
#pragma unroll
    for (int r = 0; r < 8; ++r)
      tmax[r] = fmaxf(fmaxf(l[0][r], l[1][r]), fmaxf(l[2][r], l[3][r]));
#pragma unroll
    for (int off = 1; off < 16; off <<= 1)
#pragma unroll
      for (int r = 0; r < 8; ++r)
        tmax[r] = fmaxf(tmax[r], __shfl_xor(tmax[r], off, 32));

    float corr[8];
#pragma unroll
    for (int r = 0; r < 8; ++r) {
      const float nm = fmaxf(rmax[r], tmax[r]);
      corr[r] = __expf(rmax[r] - nm);
      rmax[r] = nm;
    }

    // P = exp(l - rmax), staged to LDS (same-wave DS is in-order)
#pragma unroll
    for (int t = 0; t < 4; ++t)
#pragma unroll
      for (int r = 0; r < 8; ++r) {
        const float p = __expf(l[t][r] - rmax[r]);
        ldsP[w][(r + 8 * hh) * 64 + t * 16 + nl] = (__bf16)p;
      }

    // ---- reload P as two 16x32 A fragments
    const v16bf ap0 = load_A_frag(ldsP[w], 64, 0, 0, lane);
    const v16bf ap1 = load_A_frag(ldsP[w], 64, 0, 32, lane);

    // ---- row sums via WMMA: P @ ones -> every column holds the row sum
    v8f s = v8f_zero();
    s = wmma_bf16(ap0, ones, s);
    s = wmma_bf16(ap1, ones, s);
#pragma unroll
    for (int r = 0; r < 8; ++r) rsum[r] = rsum[r] * corr[r] + s[r];
#pragma unroll
    for (int r = 0; r < 8; ++r) {
      o0[r] *= corr[r]; o1[r] *= corr[r]; o2[r] *= corr[r]; o3[r] *= corr[r];
    }

    // ---- O += P @ V  (vTile rows are d, contiguous in local kpos => NT B)
    o0 = wmma_bf16(ap0, load_B_frag(vTile[buf], 64, 0, 0, lane), o0);
    o0 = wmma_bf16(ap1, load_B_frag(vTile[buf], 64, 0, 32, lane), o0);
    o1 = wmma_bf16(ap0, load_B_frag(vTile[buf], 64, 16, 0, lane), o1);
    o1 = wmma_bf16(ap1, load_B_frag(vTile[buf], 64, 16, 32, lane), o1);
    o2 = wmma_bf16(ap0, load_B_frag(vTile[buf], 64, 32, 0, lane), o2);
    o2 = wmma_bf16(ap1, load_B_frag(vTile[buf], 64, 32, 32, lane), o2);
    o3 = wmma_bf16(ap0, load_B_frag(vTile[buf], 64, 48, 0, lane), o3);
    o3 = wmma_bf16(ap1, load_B_frag(vTile[buf], 64, 48, 32, lane), o3);

    buf ^= 1;
  }

  // ---- normalize and store bf16 head-interleaved [b, q, h*64 + d]
  const size_t rowbase = ((size_t)(b * QL + qt * 16)) * E_DIM + h * HD;
#pragma unroll
  for (int r = 0; r < 8; ++r) {
    const float inv = 1.0f / rsum[r];
    const size_t ro = rowbase + (size_t)(r + 8 * hh) * E_DIM;
    O[ro + 0  + nl] = (__bf16)(o0[r] * inv);
    O[ro + 16 + nl] = (__bf16)(o1[r] * inv);
    O[ro + 32 + nl] = (__bf16)(o2[r] * inv);
    O[ro + 48 + nl] = (__bf16)(o3[r] * inv);
  }
}

// ------------------------------------------------------------------- launch
extern "C" void kernel_launch(void* const* d_in, const int* in_sizes, int n_in,
                              void* d_out, int out_size, void* d_ws, size_t ws_size,
                              hipStream_t stream) {
  const float* q_f  = (const float*)d_in[0];
  const float* k_f  = (const float*)d_in[1];
  const float* v_f  = (const float*)d_in[2];
  const int*   mask = (const int*)d_in[3];
  const float* wq_f = (const float*)d_in[4];
  const float* wk_f = (const float*)d_in[5];
  const float* wv_f = (const float*)d_in[6];
  const float* wo_f = (const float*)d_in[7];

  char* ws = (char*)d_ws;
  const size_t MB = (size_t)1 << 20;
  __bf16* Wq = (__bf16*)(ws + 0 * MB);    // 4 x 2MB weights
  __bf16* Wk = (__bf16*)(ws + 2 * MB);
  __bf16* Wv = (__bf16*)(ws + 4 * MB);
  __bf16* Wo = (__bf16*)(ws + 6 * MB);
  __bf16* Xq = (__bf16*)(ws + 8 * MB);    // bf16 activations, 8MB each
  __bf16* Xk = (__bf16*)(ws + 16 * MB);
  __bf16* Xv = (__bf16*)(ws + 24 * MB);
  __bf16* Qb = (__bf16*)(ws + 32 * MB);   // projected Q
  __bf16* Kb = (__bf16*)(ws + 40 * MB);   // projected K
  __bf16* Vt = (__bf16*)(ws + 48 * MB);   // projected V, [b,h][d][k]
  __bf16* Ob = (__bf16*)(ws + 56 * MB);   // attention output (total 64MB)

  const int nIn4 = BB * QL * E_DIM / 4;   // 1M float4
  const int nW4  = E_DIM * E_DIM / 4;     // 256K float4
  cvt_f32_to_bf16<<<nIn4 / 256, 256, 0, stream>>>(q_f, Xq, nIn4);
  cvt_f32_to_bf16<<<nIn4 / 256, 256, 0, stream>>>(k_f, Xk, nIn4);
  cvt_f32_to_bf16<<<nIn4 / 256, 256, 0, stream>>>(v_f, Xv, nIn4);
  cvt_f32_to_bf16<<<nW4 / 256, 256, 0, stream>>>(wq_f, Wq, nW4);
  cvt_f32_to_bf16<<<nW4 / 256, 256, 0, stream>>>(wk_f, Wk, nW4);
  cvt_f32_to_bf16<<<nW4 / 256, 256, 0, stream>>>(wv_f, Wv, nW4);
  cvt_f32_to_bf16<<<nW4 / 256, 256, 0, stream>>>(wo_f, Wo, nW4);

  const dim3 gg(E_DIM / 64, (BB * QL) / 128);  // (16, 32)
  gemm_nt_bf16<1><<<gg, 128, 0, stream>>>(Xq, Wq, Qb, BB * QL, E_DIM, E_DIM);
  gemm_nt_bf16<1><<<gg, 128, 0, stream>>>(Xk, Wk, Kb, BB * QL, E_DIM, E_DIM);
  gemm_nt_bf16<2><<<gg, 128, 0, stream>>>(Xv, Wv, Vt, BB * QL, E_DIM, E_DIM);

  // 512 blocks x 8 waves: one block per (b, h, 128 q rows)
  attn_fused<<<BB * HEADS * (QL / 128), 256, 0, stream>>>(Qb, Kb, Vt, mask, Ob);

  gemm_nt_bf16<0><<<gg, 128, 0, stream>>>(Ob, Wo, (float*)d_out, BB * QL, E_DIM,
                                          E_DIM);
}
